// BiDAF_7069516169665
// MI455X (gfx1250) — matrix operand order repaired and checked by reading, not compiled, not run on hardware
//
#include <hip/hip_runtime.h>
#include <hip/hip_bf16.h>

// ---------------------------------------------------------------------------
// BiDAF embedding encoder for MI455X (gfx1250), wave32 + WMMA bf16.
//   dims: H=356 (pad 384), 2H=712 (pad 736), 4H=1424 (89x16), convK=500 (pad 512)
// ---------------------------------------------------------------------------

typedef __bf16 bf16;
typedef __attribute__((ext_vector_type(16))) __bf16 v16bf;
typedef __attribute__((ext_vector_type(8)))  __bf16 v8bf;
typedef __attribute__((ext_vector_type(8)))  float  v8f;

#define H_DIM 356
#define HP    384
#define H2    712
#define H2P   736
#define G4    1424          // 4*H, 89 tiles of 16
#define NTOK  9216
#define NCTX  8192
#define NQ    1024
#define WCH   16
#define KC    512           // padded conv im2col K (5*100 -> 512)
#define NFP   112           // padded filter count (100 -> 112)

__device__ __forceinline__ bf16 f2bf(float x) {
  unsigned u = __builtin_bit_cast(unsigned, x);
  unsigned r = u + 0x7fffu + ((u >> 16) & 1u);           // round-nearest-even
  unsigned short h = (unsigned short)(r >> 16);
  return __builtin_bit_cast(bf16, h);
}
__device__ __forceinline__ float sigm(float x) { return 1.0f / (1.0f + __expf(-x)); }

__device__ __forceinline__ v8f wmma_bf16(v16bf a, v16bf b, v8f c) {
  return __builtin_amdgcn_wmma_f32_16x16x32_bf16(false, a, false, b, (short)0, c,
                                                 false, false);
}

// A fragment (16x32 bf16): lane holds row m=lane&15; K chunk
// k0 + (lane>>4)*16 .. +15 contiguous (two 16B chunks), matching B.
__device__ __forceinline__ v16bf load_a_frag(const bf16* A, int lda, int k0, int lane) {
  const bf16* p = A + (size_t)(lane & 15) * lda + k0 + ((lane >> 4) << 4);
  v8bf lo = *(const v8bf*)p;
  v8bf hi = *(const v8bf*)(p + 8);
  v16bf r;
#pragma unroll
  for (int i = 0; i < 8; ++i) { r[i] = lo[i]; r[i + 8] = hi[i]; }
  return r;
}

// B fragment (32x16 bf16) from W stored row-major [n][k] (i.e. B = W^T):
// lane holds col n=n0+(lane&15), K chunk k0+(lane>>4)*16 .. +15 contiguous.
__device__ __forceinline__ v16bf load_b_frag(const bf16* W, int ldw, int n0, int k0,
                                             int lane) {
  const bf16* p = W + (size_t)(n0 + (lane & 15)) * ldw + k0 + ((lane >> 4) << 4);
  return *(const v16bf*)p;
}

// ------------------------------ prep kernels -------------------------------

__global__ void k_cvt_pad(const float* __restrict__ src, bf16* __restrict__ dst,
                          int srows, int scols, int prows, int dstride) {
  int idx = blockIdx.x * 256 + threadIdx.x;
  int total = prows * dstride;
  if (idx >= total) return;
  int r = idx / dstride, c = idx - r * dstride;
  float v = (r < srows && c < scols) ? src[(size_t)r * scols + c] : 0.0f;
  dst[idx] = f2bf(v);
}

// conv_w [100,100,5] -> Wc [112][512] with Wc[f][k*100+c] = conv_w[f][c][k]
__global__ void k_conv_w(const float* __restrict__ cw, bf16* __restrict__ dst) {
  int idx = blockIdx.x * 256 + threadIdx.x;
  if (idx >= NFP * KC) return;
  int f = idx / KC, j = idx - f * KC;
  float v = 0.0f;
  if (f < 100 && j < 500) {
    int k = j / 100, c = j - k * 100;
    v = cw[(size_t)f * 500 + c * 5 + k];
  }
  dst[idx] = f2bf(v);
}

// char_emb [96,100] -> bf16 [96][112]
__global__ void k_ce(const float* __restrict__ s, bf16* __restrict__ d) {
  int idx = blockIdx.x * 256 + threadIdx.x;
  if (idx >= 96 * 112) return;
  int r = idx / 112, c = idx - r * 112;
  d[idx] = f2bf(c < 100 ? s[(size_t)r * 100 + c] : 0.0f);
}

__global__ void k_addb(const float* __restrict__ a, const float* __restrict__ b,
                       float* __restrict__ o, int n) {
  int idx = blockIdx.x * 256 + threadIdx.x;
  if (idx < n) o[idx] = a[idx] + b[idx];
}

// zero the pad columns 712..735 of the layer-1 activation buffer
__global__ void k_padcols(bf16* __restrict__ a1) {
  int idx = blockIdx.x * 256 + threadIdx.x;
  if (idx >= NTOK * 24) return;
  int r = idx / 24, c = idx - r * 24;
  a1[(size_t)r * H2P + 712 + c] = f2bf(0.0f);
}

// -------------------- char embed + conv(WMMA) + maxpool --------------------
// One token per iteration: im2col patch [16][512] in LDS, GEMM vs Wc [112][512],
// maxpool over the 16 positions via v8f reduce + shfl_xor(16).

__global__ void __launch_bounds__(256) k_feats(
    const int* __restrict__ ctx_ids, const int* __restrict__ q_ids,
    const float* __restrict__ ctx_elmo, const float* __restrict__ q_elmo,
    const float* __restrict__ conv_b, const bf16* __restrict__ ce_g,
    const bf16* __restrict__ Wc, float* __restrict__ act, bf16* __restrict__ act_bf) {
  __shared__ __align__(16) bf16 ce[96 * 112];
  __shared__ __align__(32) bf16 patch[16 * KC];
  __shared__ int ids[WCH];
  int tid = threadIdx.x, lane = tid & 31, w = tid >> 5;

  for (int i = tid; i < 96 * 112; i += 256) ce[i] = ce_g[i];

  for (int tok = blockIdx.x; tok < NTOK; tok += gridDim.x) {
    const int* idsrc = (tok < NCTX) ? (ctx_ids + (size_t)tok * WCH)
                                    : (q_ids + (size_t)(tok - NCTX) * WCH);
    if (tid < WCH) ids[tid] = idsrc[tid];
    __syncthreads();  // ids ready; prior iteration's WMMA reads of patch done

    for (int i = tid; i < 16 * KC; i += 256) {
      int m = i >> 9, j = i & (KC - 1);
      bf16 v = f2bf(0.0f);
      if (j < 500) {
        int k = j / 100, c = j - k * 100;
        int p = m + k - 2;                          // same-padding window
        if (p >= 0 && p < WCH) v = ce[ids[p] * 112 + c];
      }
      patch[i] = v;
    }
    __syncthreads();

    // elmo columns + bf16 pad columns
    const float* esrc = (tok < NCTX) ? (ctx_elmo + (size_t)tok * 256)
                                     : (q_elmo + (size_t)(tok - NCTX) * 256);
    for (int j = tid; j < 284; j += 256) {
      if (j < 256) {
        float e = esrc[j];
        act[(size_t)tok * H_DIM + 100 + j] = e;
        act_bf[(size_t)tok * HP + 100 + j] = f2bf(e);
      } else {
        act_bf[(size_t)tok * HP + 356 + (j - 256)] = f2bf(0.0f);
      }
    }

    if (w < 7) {  // 7 filter tiles of 16 across waves 0..6
      v8f acc = {};
#pragma unroll
      for (int kt = 0; kt < KC / 32; ++kt) {
        v16bf a = load_a_frag(patch, KC, kt * 32, lane);
        v16bf b = load_b_frag(Wc, KC, w * 16, kt * 32, lane);
        acc = wmma_bf16(a, b, acc);
      }
      float pm = acc[0];
#pragma unroll
      for (int v = 1; v < 8; ++v) pm = fmaxf(pm, acc[v]);      // max over 8 positions
      pm = fmaxf(pm, __shfl_xor(pm, 16, 32));                  // other 8 positions
      int f = w * 16 + (lane & 15);
      if (lane < 16 && f < 100) {
        float val = pm + conv_b[f];
        act[(size_t)tok * H_DIM + f] = val;
        act_bf[(size_t)tok * HP + f] = f2bf(val);
      }
    }
  }
}

// ------------------------------ highway layer ------------------------------
// In-place: x = sigmoid(xWg+gb) * relu(xWp+pb) + (1-g) * x. Block owns 32 rows,
// stages bf16 A tile in LDS; each wave fuses both GEMMs per 16x16 output tile.

__global__ void __launch_bounds__(256) k_highway(
    float* __restrict__ act, bf16* __restrict__ act_bf,
    const bf16* __restrict__ Wp, const float* __restrict__ pb,
    const bf16* __restrict__ Wg, const float* __restrict__ gb) {
  __shared__ __align__(32) bf16 As[32 * HP];
  int tid = threadIdx.x, lane = tid & 31, w = tid >> 5;
  int row0 = blockIdx.x * 32;

  const uint4* src = (const uint4*)(act_bf + (size_t)row0 * HP);
  uint4* dst = (uint4*)As;
  for (int i = tid; i < 32 * HP * 2 / 16; i += 256) dst[i] = src[i];
  __syncthreads();

  int mt = w & 1;
  const bf16* Am = As + mt * 16 * HP;
  for (int nt = (w >> 1); nt < 23; nt += 4) {   // 368 cols covered, 356 valid
    v8f ap = {}, ag = {};
#pragma unroll
    for (int kt = 0; kt < HP / 32; ++kt) {
      v16bf a = load_a_frag(Am, HP, kt * 32, lane);
      v16bf bp = load_b_frag(Wp, HP, nt * 16, kt * 32, lane);
      ap = wmma_bf16(a, bp, ap);
      v16bf bg = load_b_frag(Wg, HP, nt * 16, kt * 32, lane);
      ag = wmma_bf16(a, bg, ag);
    }
    int n = nt * 16 + (lane & 15);
    if (n < H_DIM) {
      float pbv = pb[n], gbv = gb[n];
#pragma unroll
      for (int v = 0; v < 8; ++v) {
        int m = row0 + mt * 16 + v + ((lane >> 4) << 3);
        float p = fmaxf(ap[v] + pbv, 0.0f);
        float g = sigm(ag[v] + gbv);
        float xo = act[(size_t)m * H_DIM + n];     // each element owned by one lane
        float xn = g * p + (1.0f - g) * xo;
        act[(size_t)m * H_DIM + n] = xn;
        act_bf[(size_t)m * HP + n] = f2bf(xn);
      }
    }
  }
}

// ------------------- LSTM input projection xg = x Wih^T + b ----------------
// grid (NTOK/32, 2 dirs). Dynamic LDS A tile [32][Kp].

__global__ void __launch_bounds__(256) k_xgemm(
    const bf16* __restrict__ act_bf, int Kp, int ktiles,
    const bf16* __restrict__ W, const float* __restrict__ bsum,
    float* __restrict__ xg) {
  extern __shared__ v8f smem_v[];                    // 32B-aligned dynamic LDS
  bf16* As = (bf16*)smem_v;
  int tid = threadIdx.x, lane = tid & 31, w = tid >> 5;
  int row0 = blockIdx.x * 32;
  int dir = blockIdx.y;
  const bf16* Wd = W + (size_t)dir * G4 * Kp;
  const float* bd = bsum + dir * G4;

  const uint4* src = (const uint4*)(act_bf + (size_t)row0 * Kp);
  uint4* dst = (uint4*)As;
  int nvec = 32 * Kp * 2 / 16;
  for (int i = tid; i < nvec; i += 256) dst[i] = src[i];
  __syncthreads();

  int mt = w & 1;
  const bf16* Am = As + mt * 16 * Kp;
  for (int nt = (w >> 1); nt < 89; nt += 4) {
    v8f acc = {};
    for (int kt = 0; kt < ktiles; ++kt) {
      v16bf a = load_a_frag(Am, Kp, kt * 32, lane);
      v16bf b = load_b_frag(Wd, Kp, nt * 16, kt * 32, lane);
      acc = wmma_bf16(a, b, acc);
    }
    int n = nt * 16 + (lane & 15);
    float bias = bd[n];
#pragma unroll
    for (int v = 0; v < 8; ++v) {
      int m = row0 + mt * 16 + v + ((lane >> 4) << 3);
      xg[((size_t)dir * NTOK + m) * G4 + n] = acc[v] + bias;
    }
  }
}

// ----------------------------- LSTM recurrence -----------------------------
// 4 blocks = (doc, dir). Per step: gates = Whh*h via WMMA matvec (h replicated
// as B column 0, W rows streamed from L2 as A fragments), then fused cell.

__global__ void __launch_bounds__(512) k_recur(
    const float* __restrict__ xg, const bf16* __restrict__ Whh,
    const float* __restrict__ h0_all, const float* __restrict__ c0_all,
    int layer, bf16* __restrict__ out_bf, float* __restrict__ out_f32) {
  __shared__ __align__(32) bf16 h_bf[HP];
  __shared__ __align__(32) float gates[G4];
  __shared__ float cst[H_DIM];
  int tid = threadIdx.x, lane = tid & 31, w = tid >> 5;
  int doc = blockIdx.x >> 1, dir = blockIdx.x & 1;
  int T = doc ? NQ : NCTX;
  int rowbase = doc ? NCTX : 0;
  size_t outbase = doc ? (size_t)NCTX * H2 : 0;
  int coff = dir * H_DIM;
  const bf16* Wd = Whh + (size_t)dir * G4 * HP;
  const float* h0 = h0_all + ((size_t)layer * 2 + dir) * H_DIM;
  const float* c0 = c0_all + ((size_t)layer * 2 + dir) * H_DIM;

  for (int u = tid; u < HP; u += 512) {
    h_bf[u] = f2bf(u < H_DIM ? h0[u] : 0.0f);
    if (u < H_DIM) cst[u] = c0[u];
  }
  __syncthreads();

  for (int s = 0; s < T; ++s) {
    int trow = dir ? (T - 1 - s) : s;
    const float* xr = xg + ((size_t)dir * NTOK + rowbase + trow) * G4;

    for (int tile = w; tile < 89; tile += 16) {
      v8f acc = {};
      const bf16* Wt = Wd + (size_t)tile * 16 * HP;
#pragma unroll
      for (int kt = 0; kt < HP / 32; ++kt) {
        v16bf a = load_a_frag(Wt, HP, kt * 32, lane);
        v16bf b = {};
        if ((lane & 15) == 0)   // column 0 of B carries h, rest are zero
          b = *(const v16bf*)(h_bf + kt * 32 + ((lane >> 4) << 4));
        acc = wmma_bf16(a, b, acc);
      }
      if ((lane & 15) == 0)     // lanes 0/16 own column 0 of D
        *(v8f*)(gates + tile * 16 + ((lane >> 4) << 3)) = acc;
    }
    __syncthreads();

    if (tid < H_DIM) {
      int u = tid;
      float ig = sigm(gates[u] + xr[u]);
      float fg = sigm(gates[H_DIM + u] + xr[H_DIM + u]);
      float gg = tanhf(gates[2 * H_DIM + u] + xr[2 * H_DIM + u]);
      float og = sigm(gates[3 * H_DIM + u] + xr[3 * H_DIM + u]);
      float c = fg * cst[u] + ig * gg;
      float h = og * tanhf(c);
      cst[u] = c;
      h_bf[u] = f2bf(h);
      if (layer == 0)
        out_bf[(size_t)(rowbase + trow) * H2P + coff + u] = f2bf(h);
      else
        out_f32[outbase + (size_t)trow * H2 + coff + u] = h;
    } else if (tid >= 512 - 89 && s + 1 < T) {
      // prefetch next timestep's xg row (89 x 64B lines) -> global_prefetch_b8
      int j = tid - (512 - 89);
      int tn = dir ? (T - 2 - s) : s + 1;
      __builtin_prefetch(xg + ((size_t)dir * NTOK + rowbase + tn) * G4 + j * 16, 0, 0);
    }
    __syncthreads();
  }
}

// --------------------------------- host ------------------------------------

extern "C" void kernel_launch(void* const* d_in, const int* in_sizes, int n_in,
                              void* d_out, int out_size, void* d_ws, size_t ws_size,
                              hipStream_t stream) {
  (void)in_sizes; (void)n_in; (void)out_size; (void)ws_size;

  const int*   ctx_ids  = (const int*)d_in[0];
  const int*   q_ids    = (const int*)d_in[1];
  const float* ctx_elmo = (const float*)d_in[2];
  const float* q_elmo   = (const float*)d_in[3];
  const float* char_emb = (const float*)d_in[4];
  const float* conv_w   = (const float*)d_in[5];
  const float* conv_b   = (const float*)d_in[6];
  const float* hw_pw    = (const float*)d_in[7];
  const float* hw_pb    = (const float*)d_in[8];
  const float* hw_gw    = (const float*)d_in[9];
  const float* hw_gb    = (const float*)d_in[10];
  const float* w_ih0    = (const float*)d_in[11];
  const float* w_hh0    = (const float*)d_in[12];
  const float* b_ih0    = (const float*)d_in[13];
  const float* b_hh0    = (const float*)d_in[14];
  const float* w_ih1    = (const float*)d_in[15];
  const float* w_hh1    = (const float*)d_in[16];
  const float* b_ih1    = (const float*)d_in[17];
  const float* b_hh1    = (const float*)d_in[18];
  const float* h0       = (const float*)d_in[19];
  const float* c0       = (const float*)d_in[20];

  char* ws = (char*)d_ws;
  size_t off = 0;
  auto alloc = [&](size_t bytes) -> char* {
    off = (off + 255) & ~(size_t)255;
    char* p = ws + off; off += bytes; return p;
  };
  float* act     = (float*)alloc((size_t)NTOK * H_DIM * 4);
  bf16*  act_bf  = (bf16*) alloc((size_t)NTOK * HP * 2);
  bf16*  act1_bf = (bf16*) alloc((size_t)NTOK * H2P * 2);
  float* xg      = (float*)alloc(2ULL * NTOK * G4 * 4);
  bf16*  Wc      = (bf16*) alloc((size_t)NFP * KC * 2);
  bf16*  ceb     = (bf16*) alloc(96 * 112 * 2);
  bf16*  hwp     = (bf16*) alloc(2ULL * HP * HP * 2);
  bf16*  hwg     = (bf16*) alloc(2ULL * HP * HP * 2);
  bf16*  wih0    = (bf16*) alloc(2ULL * G4 * HP * 2);
  bf16*  whh0    = (bf16*) alloc(2ULL * G4 * HP * 2);
  bf16*  wih1    = (bf16*) alloc(2ULL * G4 * H2P * 2);
  bf16*  whh1    = (bf16*) alloc(2ULL * G4 * HP * 2);
  float* bs0     = (float*)alloc(2 * G4 * 4);
  float* bs1     = (float*)alloc(2 * G4 * 4);

  auto cvt = [&](const float* s, bf16* d, int sr, int sc, int pr, int ds_) {
    int total = pr * ds_;
    k_cvt_pad<<<(total + 255) / 256, 256, 0, stream>>>(s, d, sr, sc, pr, ds_);
  };
  // weight conversion / padding (rows padded with zeros so pad tiles are exact)
  cvt(hw_pw,             hwp,           356, 356, HP, HP);
  cvt(hw_pw + 356 * 356, hwp + HP * HP, 356, 356, HP, HP);
  cvt(hw_gw,             hwg,           356, 356, HP, HP);
  cvt(hw_gw + 356 * 356, hwg + HP * HP, 356, 356, HP, HP);
  cvt(w_ih0, wih0, 2 * G4, 356, 2 * G4, HP);
  cvt(w_hh0, whh0, 2 * G4, 356, 2 * G4, HP);
  cvt(w_ih1, wih1, 2 * G4, 712, 2 * G4, H2P);
  cvt(w_hh1, whh1, 2 * G4, 356, 2 * G4, HP);
  k_conv_w<<<(NFP * KC + 255) / 256, 256, 0, stream>>>(conv_w, Wc);
  k_ce<<<(96 * 112 + 255) / 256, 256, 0, stream>>>(char_emb, ceb);
  k_addb<<<(2 * G4 + 255) / 256, 256, 0, stream>>>(b_ih0, b_hh0, bs0, 2 * G4);
  k_addb<<<(2 * G4 + 255) / 256, 256, 0, stream>>>(b_ih1, b_hh1, bs1, 2 * G4);
  k_padcols<<<(NTOK * 24 + 255) / 256, 256, 0, stream>>>(act1_bf);

  // char-CNN + elmo concat
  k_feats<<<256, 256, 0, stream>>>(ctx_ids, q_ids, ctx_elmo, q_elmo, conv_b,
                                   ceb, Wc, act, act_bf);
  // highway x2 (in place)
  k_highway<<<NTOK / 32, 256, 0, stream>>>(act, act_bf, hwp, hw_pb, hwg, hw_gb);
  k_highway<<<NTOK / 32, 256, 0, stream>>>(act, act_bf, hwp + HP * HP, hw_pb + 356,
                                           hwg + HP * HP, hw_gb + 356);
  // BiLSTM layer 0
  dim3 gg(NTOK / 32, 2);
  k_xgemm<<<gg, 256, 32 * HP * 2, stream>>>(act_bf, HP, HP / 32, wih0, bs0, xg);
  k_recur<<<4, 512, 0, stream>>>(xg, whh0, h0, c0, 0, act1_bf, nullptr);
  // BiLSTM layer 1 (xg buffer reused)
  k_xgemm<<<gg, 256, 32 * H2P * 2, stream>>>(act1_bf, H2P, H2P / 32, wih1, bs1, xg);
  k_recur<<<4, 512, 0, stream>>>(xg, whh1, h0, c0, 1, nullptr, (float*)d_out);
}
